// AttendModule_67611375173861
// MI455X (gfx1250) — compile-verified, matrix operand order checked by reading, not run on hardware
//
#include <hip/hip_runtime.h>
#include <stdint.h>

// ---------------------------------------------------------------------------
// AttendModule for MI455X (gfx1250, wave32, WMMA bf16 16x16x32)
// B=4, C=256, C2=128, T=16, S=256, N=T*S=4096
// Stage 0: bf16 weight conversion (Wq pre-scaled by 1/sqrt(128))
// Stage 1: x (b,c,n) -> Xt (b,n,c) bf16 via LDS tile transpose
// Stage 2: WMMA projections: Qt,Mt as (b,n,c2) bf16 ; V as (b,c2,n) fp32+bf16
// Stage 3: fused attention, block = 4 waves sharing one (b, key-frame tk):
//          M frame async-staged to LDS -> logits via WMMA -> 2-pass softmax
//          -> V frame async-staged into same LDS -> P x V^T via WMMA
// ---------------------------------------------------------------------------

typedef __bf16 bf16_t;
typedef bf16_t v16bf __attribute__((ext_vector_type(16)));
typedef float  v8f   __attribute__((ext_vector_type(8)));

#define B_   4
#define C_   256
#define C2_  128
#define T_   16
#define S_   256
#define N_   4096
#define LSTRIDE 264    // logits row stride (floats): 16B aligned, bank-staggered
#define MSTRIDE 136    // staged M row stride (bf16):  272B, 16B aligned
#define VSTRIDE 264    // staged V row stride (bf16):  528B, 16B aligned

union Frag16 { v16bf v; uint16_t u[16]; uint32_t d[8]; uint4 q[2]; };

// native converts (v_cvt_pk_bf16_f32 on gfx1250 - confirmed in disasm)
__device__ __forceinline__ uint32_t pk_bf16(float a, float b) {
    union { bf16_t h[2]; uint32_t u; } c;
    c.h[0] = (bf16_t)a; c.h[1] = (bf16_t)b;
    return c.u;
}
__device__ __forceinline__ uint16_t bf16_bits(float a) {
    union { bf16_t h; uint16_t u; } c;
    c.h = (bf16_t)a;
    return c.u;
}

// Load a 16x32 (A) / 32x16 (B) bf16 fragment slice for this lane.
// Per ISA layout: halves 0-7 <- K = k0 + 8*half + i ; halves 8-15 <- K + 16.
// Works for both global (global_load_b128) and LDS (ds_load_b128) pointers.
__device__ __forceinline__ v16bf load_frag(const uint16_t* row, int k0, int half) {
    Frag16 f;
    f.q[0] = *(const uint4*)(row + k0 + 8 * half);
    f.q[1] = *(const uint4*)(row + k0 + 16 + 8 * half);
    return f.v;
}

__device__ __forceinline__ v8f wmma_bf16(v16bf a, v16bf b, v8f c) {
    return __builtin_amdgcn_wmma_f32_16x16x32_bf16(false, a, false, b,
                                                   (short)0, c, false, false);
}

// CDNA5 async copy global -> LDS (16B per lane), tracked by ASYNCcnt
__device__ __forceinline__ void async_ld_b128(uint32_t lds_off, uint64_t gaddr) {
    asm volatile("global_load_async_to_lds_b128 %0, %1, off"
                 :: "v"(lds_off), "v"(gaddr) : "memory");
}
__device__ __forceinline__ void wait_async0() {
    asm volatile("s_wait_asynccnt 0x0" ::: "memory");
}

// ---------------- Stage 0: weights -> bf16 (Wq scaled by 1/sqrt(C2)) -------
__global__ void k_prep_w(const float* __restrict__ Wq, const float* __restrict__ Wm,
                         const float* __restrict__ Wv,
                         uint32_t* __restrict__ Wqb, uint32_t* __restrict__ Wmb,
                         uint32_t* __restrict__ Wvb) {
    int i = blockIdx.x * 256 + threadIdx.x;   // pair index
    if (i < C2_ * C_ / 2) {
        const float s = 0.08838834764831845f;  // 1/sqrt(128)
        Wqb[i] = pk_bf16(Wq[2 * i] * s, Wq[2 * i + 1] * s);
        Wmb[i] = pk_bf16(Wm[2 * i], Wm[2 * i + 1]);
        Wvb[i] = pk_bf16(Wv[2 * i], Wv[2 * i + 1]);
    }
}

// ---------------- Stage 1: x (b,c,n) fp32 -> Xt (b,n,c) bf16 ---------------
__global__ void k_transpose_x(const float* __restrict__ x, uint16_t* __restrict__ Xt) {
    __shared__ float tile[32][33];
    int b = blockIdx.z, c0 = blockIdx.y * 32, n0 = blockIdx.x * 32;
    int tn = threadIdx.x & 31;
    int tc = threadIdx.x >> 5;  // 0..7
    const float* xp = x + ((size_t)b * C_ + c0) * N_ + n0;
#pragma unroll
    for (int r = 0; r < 4; ++r) {
        int c = tc + 8 * r;
        tile[c][tn] = xp[(size_t)c * N_ + tn];
    }
    __syncthreads();
    int on  = threadIdx.x >> 3;        // 0..31
    int oc4 = (threadIdx.x & 7) * 4;   // 0..28 step 4
    uint32_t tmp[2];
    tmp[0] = pk_bf16(tile[oc4 + 0][on], tile[oc4 + 1][on]);
    tmp[1] = pk_bf16(tile[oc4 + 2][on], tile[oc4 + 3][on]);
    uint16_t* op = Xt + ((size_t)b * N_ + n0 + on) * C_ + c0 + oc4;
    *(uint2*)op = *(const uint2*)tmp;
}

// ---------------- Stage 2: projections via WMMA ----------------------------
// One wave per (b, 16-wide n-tile, o-group). Xt fragments serve as A (rows=n)
// for Qt/Mt = Xt x W^T and as B (cols=n) for V = W x Xt (identical lane maps).
// o-group og: og>>1 = {Q, M, V}, og&1 = which 4 o-tiles -> 6144 waves total.
__global__ void k_proj(const uint16_t* __restrict__ Xt,
                       const uint16_t* __restrict__ Wqb, const uint16_t* __restrict__ Wmb,
                       const uint16_t* __restrict__ Wvb,
                       uint16_t* __restrict__ Qt, uint16_t* __restrict__ Mt,
                       uint16_t* __restrict__ Vb, float* __restrict__ Vout) {
    int lane = threadIdx.x & 31;
    int gw   = blockIdx.x * (blockDim.x >> 5) + (threadIdx.x >> 5);
    int og = gw % 6;
    int nt = (gw / 6) & 255;
    int b  = gw / (6 * 256);
    int proj = og >> 1;
    int ot0  = (og & 1) * 4;
    int n0 = nt * 16;
    int half = lane >> 4, col = lane & 15;

    const uint16_t* xrow = Xt + ((size_t)b * N_ + n0 + col) * C_;
    v16bf xa[8];
#pragma unroll
    for (int cc = 0; cc < 8; ++cc) xa[cc] = load_frag(xrow, cc * 32, half);

    if (proj < 2) {
        // Qt / Mt : D rows = n, cols = c2
        const uint16_t* W = proj ? Wmb : Wqb;
        uint16_t* O = proj ? Mt : Qt;
        for (int ot = ot0; ot < ot0 + 4; ++ot) {
            v8f acc = {};
            const uint16_t* wrow = W + (size_t)(ot * 16 + col) * C_;
#pragma unroll
            for (int cc = 0; cc < 8; ++cc)
                acc = wmma_bf16(xa[cc], load_frag(wrow, cc * 32, half), acc);
            uint16_t* orow = O + ((size_t)b * N_ + n0 + 8 * half) * C2_ + ot * 16 + col;
#pragma unroll
            for (int r = 0; r < 8; ++r)
                orow[(size_t)r * C2_] = bf16_bits(acc[r]);
        }
    } else {
        // V : D rows = c2, cols = n  (fp32 to output tail + bf16 for stage 3)
        for (int ot = ot0; ot < ot0 + 4; ++ot) {
            v8f acc = {};
            const uint16_t* wrow = Wvb + (size_t)(ot * 16 + col) * C_;
#pragma unroll
            for (int cc = 0; cc < 8; ++cc)
                acc = wmma_bf16(load_frag(wrow, cc * 32, half), xa[cc], acc);
#pragma unroll
            for (int r = 0; r < 8; ++r) {
                int c2 = ot * 16 + r + 8 * half;
                size_t off = ((size_t)b * C2_ + c2) * N_ + n0 + col;
                Vout[off] = acc[r];
                Vb[off]   = bf16_bits(acc[r]);
            }
        }
    }
}

// ---------------- Stage 3: fused logits -> softmax -> PV -------------------
// Block = 128 threads = 4 waves sharing one (b, tk); wave w owns q-tile
// qt = qg*4 + w. M frame (256x128 bf16) then V frame (128x256 bf16) are
// async-staged into one shared LDS buffer; logits are per-wave fp32 in LDS.
__global__ void k_attn(const uint16_t* __restrict__ Qt, const uint16_t* __restrict__ Mt,
                       const uint16_t* __restrict__ Vb, float* __restrict__ R) {
    __shared__ __align__(16) uint16_t mvbuf[S_ * MSTRIDE];        // 69632 B
    __shared__ __align__(16) float lbuf[4][16 * LSTRIDE + 32];    // 68096 B

    int tid  = threadIdx.x;
    int lane = tid & 31;
    int w    = tid >> 5;
    int tk = blockIdx.x & 15;
    int qg = (blockIdx.x >> 4) & 63;
    int b  = blockIdx.x >> 10;
    int q0 = (qg * 4 + w) * 16;
    int half = lane >> 4, col = lane & 15;

    float* logits = lbuf[w];                 // 16 x LSTRIDE fp32
    float* maxbuf = lbuf[w] + 16 * LSTRIDE;  // 16
    float* sumbuf = maxbuf + 16;             // 16

    // ---- async-stage M frame: rows = 256 keys x 128 c (16B chunks) ----
    {
        uint32_t lds0 = (uint32_t)(uintptr_t)&mvbuf[0];
        const uint16_t* mgl = Mt + ((size_t)b * N_ + tk * S_) * C2_;
#pragma unroll
        for (int it = 0; it < 32; ++it) {
            int chunk = it * 128 + tid;           // 0..4095
            int row = chunk >> 4, cin = chunk & 15;
            async_ld_b128(lds0 + row * (MSTRIDE * 2) + cin * 16,
                          (uint64_t)(uintptr_t)(mgl + (size_t)row * C2_ + cin * 8));
        }
    }

    // Q fragments (rows = q), reused across all 16 key sub-tiles
    const uint16_t* qrow = Qt + ((size_t)b * N_ + q0 + col) * C2_;
    v16bf qa[4];
#pragma unroll
    for (int cc = 0; cc < 4; ++cc) qa[cc] = load_frag(qrow, cc * 32, half);

    wait_async0();
    __syncthreads();

    // -------- phase 1: logits + running per-row max --------
    float rmax[8];
#pragma unroll
    for (int r = 0; r < 8; ++r) rmax[r] = -3.0e38f;

    for (int j = 0; j < 16; ++j) {
        v8f acc = {};
        const uint16_t* mrow = mvbuf + (size_t)(j * 16 + col) * MSTRIDE;
#pragma unroll
        for (int cc = 0; cc < 4; ++cc)
            acc = wmma_bf16(qa[cc], load_frag(mrow, cc * 32, half), acc);
#pragma unroll
        for (int r = 0; r < 8; ++r) {
            logits[(r + 8 * half) * LSTRIDE + j * 16 + col] = acc[r];
            rmax[r] = fmaxf(rmax[r], acc[r]);
        }
    }
    // butterfly max within each 16-lane half (cols of the row)
#pragma unroll
    for (int r = 0; r < 8; ++r)
#pragma unroll
        for (int m = 1; m < 16; m <<= 1)
            rmax[r] = fmaxf(rmax[r], __shfl_xor(rmax[r], m, 32));
    if (lane == 0 || lane == 16) {
#pragma unroll
        for (int r = 0; r < 8; ++r) maxbuf[r + 8 * half] = rmax[r];
    }

    // all waves done reading M from LDS before V overwrites it
    __syncthreads();

    // ---- async-stage V frame: rows = 128 c x 256 k (16B chunks) ----
    {
        uint32_t lds0 = (uint32_t)(uintptr_t)&mvbuf[0];
        const uint16_t* vgl = Vb + (size_t)b * C2_ * N_ + tk * S_;
#pragma unroll
        for (int it = 0; it < 32; ++it) {
            int chunk = it * 128 + tid;           // 0..4095
            int row = chunk >> 5, cin = chunk & 31;
            async_ld_b128(lds0 + row * (VSTRIDE * 2) + cin * 16,
                          (uint64_t)(uintptr_t)(vgl + (size_t)row * N_ + cin * 8));
        }
    }
    wait_async0();
    __syncthreads();

    asm volatile("s_wait_dscnt 0x0" ::: "memory");
    float row_max = maxbuf[col];

    // -------- phase 2: P = exp(l - max) (bf16) ; O += P x V^T --------
    float rsum = 0.0f;
    v8f zero = {};
    v8f oacc[8];
#pragma unroll
    for (int ct = 0; ct < 8; ++ct) oacc[ct] = zero;

    const float* lrow = logits + (size_t)col * LSTRIDE;  // A-frag: lane = q row
    for (int kc = 0; kc < 8; ++kc) {
        int k0 = kc * 32;
        // 16 logits for this lane: [k0+8h .. +7] and [k0+16+8h .. +7], 32B-aligned
        float4 f0 = *(const float4*)(lrow + k0 + 8 * half);
        float4 f1 = *(const float4*)(lrow + k0 + 8 * half + 4);
        float4 f2 = *(const float4*)(lrow + k0 + 16 + 8 * half);
        float4 f3 = *(const float4*)(lrow + k0 + 16 + 8 * half + 4);
        float e[16];
        e[0]  = __expf(f0.x - row_max); e[1]  = __expf(f0.y - row_max);
        e[2]  = __expf(f0.z - row_max); e[3]  = __expf(f0.w - row_max);
        e[4]  = __expf(f1.x - row_max); e[5]  = __expf(f1.y - row_max);
        e[6]  = __expf(f1.z - row_max); e[7]  = __expf(f1.w - row_max);
        e[8]  = __expf(f2.x - row_max); e[9]  = __expf(f2.y - row_max);
        e[10] = __expf(f2.z - row_max); e[11] = __expf(f2.w - row_max);
        e[12] = __expf(f3.x - row_max); e[13] = __expf(f3.y - row_max);
        e[14] = __expf(f3.z - row_max); e[15] = __expf(f3.w - row_max);
        Frag16 p;
#pragma unroll
        for (int i = 0; i < 8; ++i) {
            rsum += e[2 * i] + e[2 * i + 1];
            p.d[i] = pk_bf16(e[2 * i], e[2 * i + 1]);
        }
#pragma unroll
        for (int ct = 0; ct < 8; ++ct) {
            const uint16_t* vrow = mvbuf + (size_t)(ct * 16 + col) * VSTRIDE;
            oacc[ct] = wmma_bf16(p.v, load_frag(vrow, k0, half), oacc[ct]);
        }
    }
    rsum += __shfl_xor(rsum, 16, 32);   // combine the two K-halves of row `col`
    if (lane < 16) sumbuf[lane] = 1.0f / rsum;
    asm volatile("s_wait_dscnt 0x0" ::: "memory");
    float scl[8];
#pragma unroll
    for (int r = 0; r < 8; ++r) scl[r] = sumbuf[r + 8 * half];

    // D layout: element r <-> q = q0 + 8*half + r (consecutive!) at column c2
#pragma unroll
    for (int ct = 0; ct < 8; ++ct) {
        int c2 = ct * 16 + col;
        float* orow = R + (((size_t)b * C2_ + c2) * T_ + tk) * N_ + q0 + 8 * half;
        float4 lo = make_float4(oacc[ct][0] * scl[0], oacc[ct][1] * scl[1],
                                oacc[ct][2] * scl[2], oacc[ct][3] * scl[3]);
        float4 hi = make_float4(oacc[ct][4] * scl[4], oacc[ct][5] * scl[5],
                                oacc[ct][6] * scl[6], oacc[ct][7] * scl[7]);
        *(float4*)orow       = lo;
        *(float4*)(orow + 4) = hi;
    }
}

// ---------------------------------------------------------------------------
extern "C" void kernel_launch(void* const* d_in, const int* in_sizes, int n_in,
                              void* d_out, int out_size, void* d_ws, size_t ws_size,
                              hipStream_t stream) {
    const float* x  = (const float*)d_in[0];
    const float* Wq = (const float*)d_in[1];
    const float* Wm = (const float*)d_in[2];
    const float* Wv = (const float*)d_in[3];

    float* R    = (float*)d_out;                          // (B,C2,Tk,Tq,H,W)
    float* Vout = R + (size_t)B_ * C2_ * T_ * N_;         // (B,C2,T,H,W)

    // workspace layout (bf16 buffers), ~20.2 MB total
    uint16_t* Wqb = (uint16_t*)d_ws;
    uint16_t* Wmb = Wqb + (size_t)C2_ * C_;
    uint16_t* Wvb = Wmb + (size_t)C2_ * C_;
    uint16_t* Xt  = Wvb + (size_t)C2_ * C_;               // (B,N,C)
    uint16_t* Qt  = Xt  + (size_t)B_ * N_ * C_;           // (B,N,C2)
    uint16_t* Mt  = Qt  + (size_t)B_ * N_ * C2_;          // (B,N,C2)
    uint16_t* Vb  = Mt  + (size_t)B_ * N_ * C2_;          // (B,C2,N)

    k_prep_w<<<(C2_ * C_ / 2 + 255) / 256, 256, 0, stream>>>(
        Wq, Wm, Wv, (uint32_t*)Wqb, (uint32_t*)Wmb, (uint32_t*)Wvb);

    dim3 tg(N_ / 32, C_ / 32, B_);
    k_transpose_x<<<tg, 256, 0, stream>>>(x, Xt);

    // 6144 waves total, 4 waves/block
    k_proj<<<1536, 128, 0, stream>>>(Xt, Wqb, Wmb, Wvb, Qt, Mt, Vb, Vout);

    // 4096 blocks x 4 waves (one (b,tk) per block), ~134.5 KB LDS each
    k_attn<<<4096, 128, 0, stream>>>(Qt, Mt, Vb, R);
}
